// CustomDecoderLayer_28698971472320
// MI455X (gfx1250) — matrix operand order repaired and checked by reading, not compiled, not run on hardware
//
#include <hip/hip_runtime.h>
#include <hip/hip_bf16.h>

typedef __attribute__((ext_vector_type(16))) __bf16 v16bf;
typedef __attribute__((ext_vector_type(8)))  float  v8f;

#define DD   1024
#define HH   16
#define DHD  64
#define MM   256
#define SSQ  2048
#define BBA  4
#define FFH  4096
#define NROW (SSQ*BBA)

// ---------------- WMMA helpers (gfx1250 bf16 16x16x32, f32 accum) -------------

__device__ __forceinline__ v8f wmma_bf(v16bf a, v16bf b, v8f c) {
  return __builtin_amdgcn_wmma_f32_16x16x32_bf16(false, a, false, b, (short)0, c, false, false);
}

// A fragment (16x32, M x K): lane l -> m = l&15, half = l>>4.
// element e<8 : k = kk + half*8 + e ; e>=8 : k = kk + 16 + half*8 + (e-8)
__device__ __forceinline__ v16bf load_afrag(const __bf16* __restrict__ A,
                                            size_t rowbase, int kk, int half) {
  const __bf16* p = A + rowbase + kk + half * 8;
  v16bf a;
#pragma unroll
  for (int e = 0; e < 8; ++e) a[e] = p[e];
#pragma unroll
  for (int e = 0; e < 8; ++e) a[8 + e] = p[16 + e];
  return a;
}

// B fragment (32x16, K x N) from row-major W (Dout x K): lane l -> n = l&15,
// half = l>>4 ; element e<8 : k = kk + half*8 + e ; e>=8 : k = kk + 16 + half*8 + (e-8)
__device__ __forceinline__ v16bf load_bfrag(const __bf16* __restrict__ W,
                                            size_t colbase, int kk, int half) {
  const __bf16* p = W + colbase + kk + half * 8;
  v16bf b;
#pragma unroll
  for (int e = 0; e < 16; ++e) b[e] = p[(e & 7) + (e >> 3) * 16];
  return b;
}

// monotonic float<->uint for atomicMax over signed floats
__device__ __forceinline__ unsigned fenc(float f) {
  unsigned u = __float_as_uint(f);
  return (u & 0x80000000u) ? ~u : (u | 0x80000000u);
}
__device__ __forceinline__ float fdec(unsigned u) {
  return (u & 0x80000000u) ? __uint_as_float(u & 0x7fffffffu) : __uint_as_float(~u);
}

// ---------------- elementwise utility kernels --------------------------------

__global__ void cvt_bf16_kernel(const float* __restrict__ in, __bf16* __restrict__ out, size_t n) {
  size_t i = (size_t)blockIdx.x * 256 + threadIdx.x;
  if (i < n) out[i] = (__bf16)in[i];
}

__global__ void zero_f32_kernel(float* __restrict__ p, size_t n) {
  size_t i = (size_t)blockIdx.x * 256 + threadIdx.x;
  if (i < n) p[i] = 0.0f;
}

// ---------------- LayerNorm: f32 in -> bf16 out ------------------------------

__global__ __launch_bounds__(256)
void ln_kernel(const float* __restrict__ x, const float* __restrict__ g,
               const float* __restrict__ b, __bf16* __restrict__ out) {
  __shared__ float s1[256], s2[256];
  const int row = blockIdx.x, tid = threadIdx.x;
  const float* xr = x + (size_t)row * DD;
  float sum = 0.f, sq = 0.f;
  for (int i = tid; i < DD; i += 256) { float v = xr[i]; sum += v; sq += v * v; }
  s1[tid] = sum; s2[tid] = sq; __syncthreads();
  for (int st = 128; st > 0; st >>= 1) {
    if (tid < st) { s1[tid] += s1[tid + st]; s2[tid] += s2[tid + st]; }
    __syncthreads();
  }
  const float mu = s1[0] * (1.0f / DD);
  const float var = s2[0] * (1.0f / DD) - mu * mu;
  const float rstd = rsqrtf(var + 1e-5f);
  __bf16* o = out + (size_t)row * DD;
  for (int i = tid; i < DD; i += 256)
    o[i] = (__bf16)((xr[i] - mu) * rstd * g[i] + b[i]);
}

// ---------------- generic bf16 WMMA GEMM: C = A @ W^T + bias -----------------
// A: N x K row-major bf16. W: Dout x K row-major bf16. block: 8 waves,
// wave tile 32x32, block tile 128x64. Epilogue: bias, relu?, +residual?,
// write f32 and/or bf16.

__global__ __launch_bounds__(256)
void gemm_bf16_kernel(const __bf16* __restrict__ A, const __bf16* __restrict__ W,
                      const float* __restrict__ bias, const float* __restrict__ residual,
                      float* __restrict__ outF, __bf16* __restrict__ outB,
                      int K, int Dout, int relu) {
  const int lane = threadIdx.x & 31, wave = threadIdx.x >> 5;
  const int l16 = lane & 15, half = lane >> 4;
  const int row0 = blockIdx.x * 128 + (wave >> 1) * 32;
  const int col0 = blockIdx.y * 64 + (wave & 1) * 32;

  v8f acc[2][2];
#pragma unroll
  for (int i = 0; i < 2; ++i)
#pragma unroll
    for (int j = 0; j < 2; ++j) acc[i][j] = (v8f){};

  const size_t arow0 = (size_t)(row0 + l16) * K;
  const size_t arow1 = (size_t)(row0 + 16 + l16) * K;
  const size_t bcol0 = (size_t)(col0 + l16) * K;
  const size_t bcol1 = (size_t)(col0 + 16 + l16) * K;

  for (int kk = 0; kk < K; kk += 32) {
    v16bf a0 = load_afrag(A, arow0, kk, half);
    v16bf a1 = load_afrag(A, arow1, kk, half);
    v16bf b0 = load_bfrag(W, bcol0, kk, half);
    v16bf b1 = load_bfrag(W, bcol1, kk, half);
    acc[0][0] = wmma_bf(a0, b0, acc[0][0]);
    acc[0][1] = wmma_bf(a0, b1, acc[0][1]);
    acc[1][0] = wmma_bf(a1, b0, acc[1][0]);
    acc[1][1] = wmma_bf(a1, b1, acc[1][1]);
  }

#pragma unroll
  for (int i = 0; i < 2; ++i)
#pragma unroll
    for (int j = 0; j < 2; ++j) {
      const int col = col0 + j * 16 + l16;
      const float bc = bias[col];
#pragma unroll
      for (int r = 0; r < 8; ++r) {
        const int row = row0 + i * 16 + r + 8 * half;
        float v = acc[i][j][r] + bc;
        if (relu) v = fmaxf(v, 0.0f);
        const size_t idx = (size_t)row * Dout + col;
        if (residual) v += residual[idx];
        if (outF) outF[idx] = v;
        if (outB) outB[idx] = (__bf16)v;
      }
    }
}

// ---------------- FAVOR+ feature map ------------------------------------------
// proj[m] = sum_d (x*DH^-1/4)[d] * feat[m][d] ; out = exp(proj - sq - stab)/16 + eps
// MODE 0: query (per-row stab), write out.
// MODE 1: key pass 1: row max -> atomicMax gmax[(b,h)].
// MODE 2: key pass 2: stab = gmax, write pk + atomicAdd pksum[(b,h),m].
// grid: (NROW/16, HH), block 256 (8 waves; wave w owns feature cols [32w,32w+32)).

template <int MODE>
__global__ __launch_bounds__(256)
void phi_kernel(const float* __restrict__ x, const __bf16* __restrict__ featbf,
                __bf16* __restrict__ out, float* __restrict__ pksum,
                unsigned* __restrict__ gmax) {
  const int h = blockIdx.y;
  const int r0 = blockIdx.x * 16;
  const int tid = threadIdx.x;
  const int lane = tid & 31, wave = tid >> 5;
  const int l16 = lane & 15, half = lane >> 4;
  const float s4 = 0.35355339059327373f;  // 64^-0.25

  __shared__ float proj[16][MM + 1];
  __shared__ float srow[16], sstab[16];

  // A fragments: 16 rows x 64 K of head h, scaled, cvt bf16 on the fly
  v16bf a[2];
#pragma unroll
  for (int t = 0; t < 2; ++t) {
    const float* p = x + (size_t)(r0 + l16) * DD + h * DHD + t * 32 + half * 8;
    v16bf av;
#pragma unroll
    for (int e = 0; e < 8; ++e) av[e] = (__bf16)(p[e] * s4);
#pragma unroll
    for (int e = 0; e < 8; ++e) av[8 + e] = (__bf16)(p[16 + e] * s4);
    a[t] = av;
  }

  const int col0 = wave * 32;
  v8f acc[2]; acc[0] = (v8f){}; acc[1] = (v8f){};
#pragma unroll
  for (int j = 0; j < 2; ++j) {
    const size_t colbase = (size_t)(col0 + j * 16 + l16) * DHD;
#pragma unroll
    for (int t = 0; t < 2; ++t) {
      v16bf bf = load_bfrag(featbf, colbase, t * 32, half);
      acc[j] = wmma_bf(a[t], bf, acc[j]);
    }
  }
#pragma unroll
  for (int j = 0; j < 2; ++j)
#pragma unroll
    for (int r = 0; r < 8; ++r)
      proj[r + 8 * half][col0 + j * 16 + l16] = acc[j][r];
  __syncthreads();

  if (tid < 16) {
    const int rr = r0 + tid;
    const float* p = x + (size_t)rr * DD + h * DHD;
    float s = 0.f;
    for (int d = 0; d < DHD; ++d) s += p[d] * p[d];
    srow[tid] = s * 0.0625f;  // 0.5 * DH^-0.5 * sum x^2
    float mx = -3e38f;
    for (int m = 0; m < MM; ++m) mx = fmaxf(mx, proj[tid][m]);
    if (MODE == 1) atomicMax(&gmax[(rr & (BBA - 1)) * HH + h], fenc(mx));
    sstab[tid] = (MODE == 2) ? fdec(gmax[(rr & (BBA - 1)) * HH + h]) : mx;
  }
  __syncthreads();

  if (MODE != 1) {
    const int row = tid >> 4;
    const int c0 = (tid & 15) * 16;
    const int rr = r0 + row;
    const float sq_ = srow[row], st = sstab[row];
    __bf16* o = out + ((size_t)rr * HH + h) * MM;
    float* ks = (MODE == 2) ? (pksum + ((size_t)(rr & (BBA - 1)) * HH + h) * MM) : nullptr;
    for (int c = c0; c < c0 + 16; ++c) {
      const float val = __expf(proj[row][c] - sq_ - st) * 0.0625f + 1e-6f;
      o[c] = (__bf16)val;
      if (MODE == 2) atomicAdd(&ks[c], val);
    }
  }
}

// ---------------- kv = sum_s pk[s,m] (x) v[s,d] per (b,h) --------------------
// grid: B*H blocks, 8 waves; wave w owns m rows [32w,32w+32), all 64 d cols.
// LDS-staged chunks of 32 s, fragments built from LDS.

__global__ __launch_bounds__(256)
void kv_kernel(const __bf16* __restrict__ pk, const float* __restrict__ vf,
               __bf16* __restrict__ kvout) {
  const int bh = blockIdx.x;
  const int b = bh >> 4, h = bh & 15;
  const int tid = threadIdx.x;
  const int lane = tid & 31, wave = tid >> 5;
  const int l16 = lane & 15, half = lane >> 4;

  __shared__ __bf16 pks[32][MM];
  __shared__ __bf16 vs[32][DHD];

  v8f acc[2][4];
#pragma unroll
  for (int i = 0; i < 2; ++i)
#pragma unroll
    for (int j = 0; j < 4; ++j) acc[i][j] = (v8f){};

  const int m0 = wave * 32;

  for (int ss = 0; ss < SSQ; ss += 32) {
    for (int i = tid; i < 32 * MM; i += 256) {
      const int sr = i >> 8, m = i & 255;
      pks[sr][m] = pk[(((size_t)(ss + sr) * BBA + b) * HH + h) * MM + m];
    }
    for (int i = tid; i < 32 * DHD; i += 256) {
      const int sr = i >> 6, d = i & 63;
      vs[sr][d] = (__bf16)vf[((size_t)(ss + sr) * BBA + b) * DD + h * DHD + d];
    }
    __syncthreads();

    v16bf af[2];
#pragma unroll
    for (int i = 0; i < 2; ++i) {
      const int m = m0 + i * 16 + l16;
      v16bf av;
#pragma unroll
      for (int e = 0; e < 8; ++e) av[e] = pks[half * 8 + e][m];
#pragma unroll
      for (int e = 0; e < 8; ++e) av[8 + e] = pks[16 + half * 8 + e][m];
      af[i] = av;
    }
    v16bf bf[4];
#pragma unroll
    for (int j = 0; j < 4; ++j) {
      const int d = j * 16 + l16;
      v16bf bv;
#pragma unroll
      for (int e = 0; e < 16; ++e) bv[e] = vs[half * 8 + (e & 7) + (e >> 3) * 16][d];
      bf[j] = bv;
    }
#pragma unroll
    for (int i = 0; i < 2; ++i)
#pragma unroll
      for (int j = 0; j < 4; ++j) acc[i][j] = wmma_bf(af[i], bf[j], acc[i][j]);
    __syncthreads();
  }

#pragma unroll
  for (int i = 0; i < 2; ++i)
#pragma unroll
    for (int j = 0; j < 4; ++j)
#pragma unroll
      for (int r = 0; r < 8; ++r) {
        const int m = m0 + i * 16 + r + 8 * half;
        const int d = j * 16 + l16;
        kvout[((size_t)bh * MM + m) * DHD + d] = (__bf16)acc[i][j][r];
      }
}

// ---------------- z[row,h] = pq[row,h,:] . pksum[b,h,:] ----------------------

__global__ __launch_bounds__(256)
void z_kernel(const __bf16* __restrict__ pq, const float* __restrict__ pksum,
              float* __restrict__ z) {
  const size_t idx = (size_t)blockIdx.x * 256 + threadIdx.x;  // < NROW*HH
  const size_t row = idx >> 4;
  const int h = (int)(idx & 15);
  const int b = (int)(row & (BBA - 1));
  const __bf16* p = pq + (row * HH + h) * MM;
  const float* ks = pksum + ((size_t)b * HH + h) * MM;
  float acc = 0.f;
  for (int m = 0; m < MM; ++m) acc += (float)p[m] * ks[m];
  z[idx] = acc;
}

// ---------------- attn out: (pq @ kv)/z -> bf16 (row, h*64+d) ----------------
// grid: (S/128, B, H); block 8 waves (4 s-groups x 2 d-groups of 32).

__global__ __launch_bounds__(256)
void attn_out_kernel(const __bf16* __restrict__ pq, const __bf16* __restrict__ kvbf,
                     const float* __restrict__ z, __bf16* __restrict__ attnbf) {
  const int b = blockIdx.y, h = blockIdx.z;
  const int tid = threadIdx.x;
  const int lane = tid & 31, wave = tid >> 5;
  const int l16 = lane & 15, half = lane >> 4;
  const int sbase = blockIdx.x * 128 + (wave >> 1) * 32;
  const int d0 = (wave & 1) * 32;
  const size_t bh = (size_t)b * HH + h;

  v8f acc[2][2];
#pragma unroll
  for (int i = 0; i < 2; ++i)
#pragma unroll
    for (int j = 0; j < 2; ++j) acc[i][j] = (v8f){};

  for (int kk = 0; kk < MM; kk += 32) {
    v16bf af[2];
#pragma unroll
    for (int i = 0; i < 2; ++i) {
      const size_t rowbase = (((size_t)(sbase + i * 16 + l16) * BBA + b) * HH + h) * MM;
      af[i] = load_afrag(pq, rowbase, kk, half);
    }
    v16bf bf[2];
#pragma unroll
    for (int j = 0; j < 2; ++j) {
      const int d = d0 + j * 16 + l16;
      v16bf bv;
#pragma unroll
      for (int e = 0; e < 16; ++e)
        bv[e] = kvbf[(bh * MM + kk + half * 8 + (e & 7) + (e >> 3) * 16) * DHD + d];
      bf[j] = bv;
    }
#pragma unroll
    for (int i = 0; i < 2; ++i)
#pragma unroll
      for (int j = 0; j < 2; ++j) acc[i][j] = wmma_bf(af[i], bf[j], acc[i][j]);
  }

#pragma unroll
  for (int i = 0; i < 2; ++i)
#pragma unroll
    for (int j = 0; j < 2; ++j)
#pragma unroll
      for (int r = 0; r < 8; ++r) {
        const int s = sbase + i * 16 + r + 8 * half;
        const int d = d0 + j * 16 + l16;
        const size_t row = (size_t)s * BBA + b;
        const float zz = z[row * HH + h];
        attnbf[row * DD + h * DHD + d] = (__bf16)(acc[i][j][r] / zz);
      }
}

// ============================= host orchestration ============================

extern "C" void kernel_launch(void* const* d_in, const int* in_sizes, int n_in,
                              void* d_out, int out_size, void* d_ws, size_t ws_size,
                              hipStream_t stream) {
  (void)in_sizes; (void)n_in; (void)out_size; (void)ws_size;
  const float* tgt    = (const float*)d_in[0];
  const float* memory = (const float*)d_in[1];
  const float* sa_w[4] = {(const float*)d_in[2], (const float*)d_in[4],
                          (const float*)d_in[6], (const float*)d_in[8]};
  const float* sa_b[4] = {(const float*)d_in[3], (const float*)d_in[5],
                          (const float*)d_in[7], (const float*)d_in[9]};
  const float* sa_feat = (const float*)d_in[10];
  const float* ca_w[4] = {(const float*)d_in[11], (const float*)d_in[13],
                          (const float*)d_in[15], (const float*)d_in[17]};
  const float* ca_b[4] = {(const float*)d_in[12], (const float*)d_in[14],
                          (const float*)d_in[16], (const float*)d_in[18]};
  const float* ca_feat = (const float*)d_in[19];
  const float* ff_w1 = (const float*)d_in[20];
  const float* ff_b1 = (const float*)d_in[21];
  const float* ff_w2 = (const float*)d_in[22];
  const float* ff_b2 = (const float*)d_in[23];
  const float* ln_g[3] = {(const float*)d_in[24], (const float*)d_in[26], (const float*)d_in[28]};
  const float* ln_b[3] = {(const float*)d_in[25], (const float*)d_in[27], (const float*)d_in[29]};

  float* tgt_cur = (float*)d_out;

  char* base = (char*)d_ws;
  size_t off = 0;
  auto alloc = [&](size_t bytes) -> void* {
    void* p = base + off;
    off = (off + bytes + 255) & ~(size_t)255;
    return p;
  };

  const size_t WSZ = (size_t)DD * DD;           // 1M
  __bf16* sa_wbf[4]; __bf16* ca_wbf[4];
  for (int i = 0; i < 4; ++i) sa_wbf[i] = (__bf16*)alloc(WSZ * 2);
  for (int i = 0; i < 4; ++i) ca_wbf[i] = (__bf16*)alloc(WSZ * 2);
  __bf16* featsa = (__bf16*)alloc((size_t)MM * DHD * 2);
  __bf16* featca = (__bf16*)alloc((size_t)MM * DHD * 2);
  __bf16* w1bf = (__bf16*)alloc((size_t)FFH * DD * 2);
  __bf16* w2bf = (__bf16*)alloc((size_t)DD * FFH * 2);
  __bf16* membf = (__bf16*)alloc((size_t)NROW * DD * 2);
  __bf16* t2bf  = (__bf16*)alloc((size_t)NROW * DD * 2);
  float*  qf = (float*)alloc((size_t)NROW * DD * 4);
  float*  kf = (float*)alloc((size_t)NROW * DD * 4);
  float*  vf = (float*)alloc((size_t)NROW * DD * 4);
  __bf16* pqbf = (__bf16*)alloc((size_t)NROW * HH * MM * 2);   // 64 MB
  __bf16* pkbf = (__bf16*)alloc((size_t)NROW * HH * MM * 2);
  float*  pksum = (float*)alloc((size_t)BBA * HH * MM * 4);
  unsigned* gmaxu = (unsigned*)alloc((size_t)BBA * HH * 4);
  __bf16* kvbf = (__bf16*)alloc((size_t)BBA * HH * MM * DHD * 2);
  float*  zbuf = (float*)alloc((size_t)NROW * HH * 4);
  __bf16* attnbf = (__bf16*)alloc((size_t)NROW * DD * 2);
  __bf16* hbuf = pqbf;  // FFN hidden (N x 4096 bf16, 64 MB) aliases pq region

  auto cvt = [&](const float* in, __bf16* out, size_t n) {
    cvt_bf16_kernel<<<dim3((unsigned)((n + 255) / 256)), 256, 0, stream>>>(in, out, n);
  };
  auto gemm = [&](const __bf16* A, const __bf16* W, const float* bias,
                  const float* residual, float* oF, __bf16* oB,
                  int K, int Dout, int relu) {
    gemm_bf16_kernel<<<dim3(NROW / 128, Dout / 64), 256, 0, stream>>>(
        A, W, bias, residual, oF, oB, K, Dout, relu);
  };

  // ---- convert weights to bf16 ----
  for (int i = 0; i < 4; ++i) cvt(sa_w[i], sa_wbf[i], WSZ);
  for (int i = 0; i < 4; ++i) cvt(ca_w[i], ca_wbf[i], WSZ);
  cvt(sa_feat, featsa, (size_t)MM * DHD);
  cvt(ca_feat, featca, (size_t)MM * DHD);
  cvt(ff_w1, w1bf, (size_t)FFH * DD);
  cvt(ff_w2, w2bf, (size_t)DD * FFH);
  cvt(memory, membf, (size_t)NROW * DD);

  // running residual in d_out
  hipMemcpyAsync(tgt_cur, tgt, (size_t)NROW * DD * sizeof(float),
                 hipMemcpyDeviceToDevice, stream);

  auto run_attn = [&](const __bf16* q_src, const __bf16* kv_src,
                      __bf16* const* wbf, const float* const* bias,
                      const __bf16* feat) {
    gemm(q_src, wbf[0], bias[0], nullptr, qf, nullptr, DD, DD, 0);   // q
    gemm(kv_src, wbf[1], bias[1], nullptr, kf, nullptr, DD, DD, 0);  // k
    gemm(kv_src, wbf[2], bias[2], nullptr, vf, nullptr, DD, DD, 0);  // v

    dim3 pg(NROW / 16, HH);
    phi_kernel<0><<<pg, 256, 0, stream>>>(qf, feat, pqbf, nullptr, nullptr);
    zero_f32_kernel<<<dim3((BBA * HH * MM + 255) / 256), 256, 0, stream>>>(
        pksum, (size_t)BBA * HH * MM);
    zero_f32_kernel<<<1, 256, 0, stream>>>((float*)gmaxu, (size_t)BBA * HH);
    phi_kernel<1><<<pg, 256, 0, stream>>>(kf, feat, pkbf, pksum, gmaxu);
    phi_kernel<2><<<pg, 256, 0, stream>>>(kf, feat, pkbf, pksum, gmaxu);

    kv_kernel<<<dim3(BBA * HH), 256, 0, stream>>>(pkbf, vf, kvbf);
    z_kernel<<<dim3((NROW * HH) / 256), 256, 0, stream>>>(pqbf, pksum, zbuf);
    attn_out_kernel<<<dim3(SSQ / 128, BBA, HH), 256, 0, stream>>>(pqbf, kvbf, zbuf, attnbf);

    // tgt += attn @ wo^T + bo
    gemm(attnbf, wbf[3], bias[3], tgt_cur, tgt_cur, nullptr, DD, DD, 0);
  };

  // ---- self-attention (pre-norm) ----
  ln_kernel<<<dim3(NROW), 256, 0, stream>>>(tgt_cur, ln_g[0], ln_b[0], t2bf);
  run_attn(t2bf, t2bf, sa_wbf, sa_b, featsa);

  // ---- cross-attention ----
  ln_kernel<<<dim3(NROW), 256, 0, stream>>>(tgt_cur, ln_g[1], ln_b[1], t2bf);
  run_attn(t2bf, membf, ca_wbf, ca_b, featca);

  // ---- FFN ----
  ln_kernel<<<dim3(NROW), 256, 0, stream>>>(tgt_cur, ln_g[2], ln_b[2], t2bf);
  gemm(t2bf, w1bf, ff_b1, nullptr, nullptr, hbuf, DD, FFH, 1);        // relu
  gemm(hbuf, w2bf, ff_b2, tgt_cur, tgt_cur, nullptr, FFH, DD, 0);     // residual
}
